// BantamAttention_29652454211594
// MI455X (gfx1250) — compile-verified, matrix operand order checked by reading, not run on hardware
//
#include <hip/hip_runtime.h>
#include <hip/hip_bf16.h>

// ---------------------------------------------------------------------------
// CDNA5 (gfx1250) sliding-window GQA attention, f16 WMMA everywhere.
// B=1, Q=1024, Dm=4096, H=32, Hkv=8, D=128, P=4096, T=5120, WINDOW=4096,
// SINKS=4 -> K_kept=4096 (sinks 0..3 + tail pos 1028..5119). Scale=1/sqrt(128).
// ---------------------------------------------------------------------------

typedef _Float16 v16h __attribute__((ext_vector_type(16)));
typedef float    v8f  __attribute__((ext_vector_type(8)));
typedef unsigned int uint4v __attribute__((ext_vector_type(4)));
typedef int      v4i  __attribute__((vector_size(16)));   // matches builtin proto

union TileU { uint4v u2[2]; v16h v; };

#define NQ      1024
#define DM      4096
#define NH      32
#define NHKV    8
#define HD      128
#define NPAST   4096
#define KKEPT   4096
#define NQKV    6144
#define QSCALE  0.08838834764831845f   // 1/sqrt(128)

// ---- CDNA5 async global->LDS copy (probe via __has_builtin) ---------------
#if defined(__has_builtin)
#  if __has_builtin(__builtin_amdgcn_global_load_async_to_lds_b128)
#    define HAVE_ASYNC_LDS 1
#  else
#    define HAVE_ASYNC_LDS 0
#  endif
#  if __has_builtin(__builtin_amdgcn_s_wait_asynccnt)
#    define HAVE_WAIT_ASYNC 1
#  else
#    define HAVE_WAIT_ASYNC 0
#  endif
#else
#  define HAVE_ASYNC_LDS 0
#  define HAVE_WAIT_ASYNC 0
#endif

#if HAVE_ASYNC_LDS
__device__ inline void async_copy16(const void* g, void* l) {
    __builtin_amdgcn_global_load_async_to_lds_b128(
        (__attribute__((address_space(1))) v4i*)g,
        (__attribute__((address_space(3))) v4i*)l, 0, 0);
}
__device__ inline void wait_async() {
#  if HAVE_WAIT_ASYNC
    __builtin_amdgcn_s_wait_asynccnt(0);
#  else
    asm volatile("s_wait_asynccnt 0" ::: "memory");
#  endif
    asm volatile("" ::: "memory");   // keep ds_loads from hoisting above wait
}
#endif

// --------------------------------------------------------------------------
// Load a 16x32 f16 tile in the documented A-matrix VGPR layout:
//   lanes 0-15 : row = rowbase + lane,     K = kbase + {0..7, 16..23}
//   lanes 16-31: row = rowbase + lane-16,  K = kbase + {8..15, 24..31}
// Works for B-tiles when the matrix is stored N-major (B^T row-major), and
// for LDS tiles (address-space inference emits ds_load_b128).
// --------------------------------------------------------------------------
__device__ inline v16h load_tile16x32(const _Float16* base,
                                      int ld, int rowbase, int kbase) {
    int lane = threadIdx.x & 31;
    int half = lane >> 4;
    int r    = lane & 15;
    const _Float16* p = base + (size_t)(rowbase + r) * ld + kbase + half * 8;
    TileU t;
    t.u2[0] = *(const uint4v*)(p);
    t.u2[1] = *(const uint4v*)(p + 16);
    return t.v;
}

// --------------------------------------------------------------------------
// Elementwise f32 -> f16 convert.
// --------------------------------------------------------------------------
__global__ void bantam_cvt_f16(const float* __restrict__ src,
                               _Float16* __restrict__ dst, int n) {
    int i = blockIdx.x * blockDim.x + threadIdx.x;
    if (i < n) dst[i] = (_Float16)src[i];
}

// Transpose-convert: src [K][N] f32 row-major -> dst [N][K] f16 row-major.
__global__ void bantam_tcvt_f16(const float* __restrict__ src,
                                _Float16* __restrict__ dst, int K, int N) {
    int i = blockIdx.x * blockDim.x + threadIdx.x;
    if (i >= K * N) return;
    int k = i / N;
    int n = i - k * N;
    dst[(size_t)n * K + k] = (_Float16)src[i];
}

// --------------------------------------------------------------------------
// WMMA GEMM: C[M][N] f32 = A[M][K] f16 @ Bt[N][K] f16 (B transposed).
// One wave computes a 32x64 tile: 2 A-tiles x 4 B-tiles -> 8 WMMAs per
// 32-deep k-step off 12 b128 loads (1.5 loads/WMMA).
// --------------------------------------------------------------------------
__global__ __launch_bounds__(32)
void bantam_gemm_wmma(const _Float16* __restrict__ A,
                      const _Float16* __restrict__ Bt,
                      float* __restrict__ C, int M, int N, int K) {
    int tn = blockIdx.x;   // N/64
    int tm = blockIdx.y;   // M/32
    v8f acc[2][4];
#pragma unroll
    for (int i = 0; i < 2; ++i)
#pragma unroll
        for (int j = 0; j < 4; ++j) acc[i][j] = (v8f){};

    for (int k = 0; k < K; k += 32) {
        v16h a0 = load_tile16x32(A, K, tm * 32,      k);
        v16h a1 = load_tile16x32(A, K, tm * 32 + 16, k);
        v16h b[4];
#pragma unroll
        for (int j = 0; j < 4; ++j)
            b[j] = load_tile16x32(Bt, K, tn * 64 + j * 16, k);
#pragma unroll
        for (int j = 0; j < 4; ++j) {
            acc[0][j] = __builtin_amdgcn_wmma_f32_16x16x32_f16(
                false, a0, false, b[j], (short)0, acc[0][j], false, false);
            acc[1][j] = __builtin_amdgcn_wmma_f32_16x16x32_f16(
                false, a1, false, b[j], (short)0, acc[1][j], false, false);
        }
    }

    int lane  = threadIdx.x & 31;
    int col   = lane & 15;
    int rbase = ((lane >> 4) << 3);
#pragma unroll
    for (int i = 0; i < 2; ++i)
#pragma unroll
        for (int j = 0; j < 4; ++j)
#pragma unroll
            for (int r = 0; r < 8; ++r)
                C[(size_t)(tm * 32 + i * 16 + rbase + r) * N
                  + tn * 64 + j * 16 + col] = acc[i][j][r];
}

// --------------------------------------------------------------------------
// RoPE + scale on Q, scatter to q16[H][Q][D].
// --------------------------------------------------------------------------
__global__ void bantam_rope_q(const float* __restrict__ qkv,
                              const float* __restrict__ cosT,
                              const float* __restrict__ sinT,
                              _Float16* __restrict__ q16) {
    int i = blockIdx.x * blockDim.x + threadIdx.x;
    if (i >= NQ * NH * HD) return;
    int d = i & 127;
    int h = (i >> 7) & 31;
    int q = i >> 12;
    int pos = NPAST + q;
    const float* row = qkv + (size_t)q * NQKV + h * HD;
    float x     = row[d];
    float other = (d < 64) ? -row[d + 64] : row[d - 64];
    float c = cosT[(size_t)pos * HD + d];
    float s = sinT[(size_t)pos * HD + d];
    q16[((size_t)h * NQ + q) * HD + d] = (_Float16)((x * c + other * s) * QSCALE);
}

// --------------------------------------------------------------------------
// Kept-K cache f16: k16[Hkv][4096][128]. kept j -> pos: j<4 ? j : 1024+j.
// --------------------------------------------------------------------------
__global__ void bantam_build_k(const float* __restrict__ qkv,
                               const float* __restrict__ past_k,
                               const float* __restrict__ cosT,
                               const float* __restrict__ sinT,
                               _Float16* __restrict__ k16) {
    int i = blockIdx.x * blockDim.x + threadIdx.x;
    if (i >= NHKV * KKEPT * HD) return;
    int d = i & 127;
    int j = (i >> 7) & 4095;
    int h = i >> 19;
    int pos = (j < 4) ? j : 1024 + j;
    float val;
    if (pos < NPAST) {
        val = past_k[((size_t)h * NPAST + pos) * HD + d];
    } else {
        int q = pos - NPAST;
        const float* row = qkv + (size_t)q * NQKV + NH * HD + h * HD;
        float x     = row[d];
        float other = (d < 64) ? -row[d + 64] : row[d - 64];
        val = x * cosT[(size_t)pos * HD + d] + other * sinT[(size_t)pos * HD + d];
    }
    k16[i] = (_Float16)val;
}

// --------------------------------------------------------------------------
// Kept-V cache TRANSPOSED f16: vt16[Hkv][128][4096] (d-major).
// --------------------------------------------------------------------------
__global__ void bantam_build_v(const float* __restrict__ qkv,
                               const float* __restrict__ past_v,
                               _Float16* __restrict__ vt16) {
    int i = blockIdx.x * blockDim.x + threadIdx.x;
    if (i >= NHKV * KKEPT * HD) return;
    int d = i & 127;
    int j = (i >> 7) & 4095;
    int h = i >> 19;
    int pos = (j < 4) ? j : 1024 + j;
    float val;
    if (pos < NPAST) {
        val = past_v[((size_t)h * NPAST + pos) * HD + d];
    } else {
        val = qkv[(size_t)(pos - NPAST) * NQKV + NH * HD + NHKV * HD + h * HD + d];
    }
    vt16[((size_t)h * HD + d) * KKEPT + j] = (_Float16)val;
}

// --------------------------------------------------------------------------
// Flash attention: one wave per (head, 16-query tile). Online softmax with
// 16-lane shfl_xor row reductions; P staged through LDS (C->A layout).
// When available, K/V tiles are double-buffered in LDS via
// GLOBAL_LOAD_ASYNC_TO_LDS_B128 + s_wait_asynccnt; else direct global loads.
// --------------------------------------------------------------------------
__global__ __launch_bounds__(32)
void bantam_attn_wmma(const _Float16* __restrict__ q16,
                      const _Float16* __restrict__ k16,
                      const _Float16* __restrict__ vt16,
                      _Float16* __restrict__ a16) {
    int h = blockIdx.x;        // q head
    int t = blockIdx.y;        // query tile
    int kvh = h >> 2;          // GQA group of 4

    const _Float16* Qb = q16  + (size_t)h   * NQ    * HD;
    const _Float16* Kb = k16  + (size_t)kvh * KKEPT * HD;
    const _Float16* Vb = vt16 + (size_t)kvh * HD    * KKEPT;

    int lane = threadIdx.x & 31;
    int half = lane >> 4;
    int l16  = lane & 15;

    v16h qa[4];
#pragma unroll
    for (int c = 0; c < 4; ++c)
        qa[c] = load_tile16x32(Qb, HD, t * 16, c * 32);

    v8f O[8];
#pragma unroll
    for (int c = 0; c < 8; ++c) O[c] = (v8f){};
    float m[8], lsum[8];
#pragma unroll
    for (int r = 0; r < 8; ++r) { m[r] = -3.0e38f; lsum[r] = 0.0f; }

    __shared__ _Float16 Plds[16 * 32];
#if HAVE_ASYNC_LDS
    __shared__ _Float16 Kl[2][32 * 128];   // [key][d], 8 KB per buffer
    __shared__ _Float16 Vl[2][128 * 32];   // [d][key], 8 KB per buffer
#endif

    int nkeys = 3088 + 16 * t;             // causal horizon
    if (nkeys > KKEPT) nkeys = KKEPT;

#if HAVE_ASYNC_LDS
    // stage tile (keys j0..j0+31) into LDS buffer `buf`
    auto stage = [&](int j0, int buf) {
        const char* gk = (const char*)(Kb + (size_t)j0 * HD);   // 8 KB contiguous
        char* lk = (char*)&Kl[buf][0];
#pragma unroll
        for (int tc = 0; tc < 16; ++tc) {
            int c = tc * 32 + lane;                 // 512 x 16B chunks
            async_copy16(gk + c * 16, lk + c * 16);
        }
        char* lv = (char*)&Vl[buf][0];
#pragma unroll
        for (int tc = 0; tc < 16; ++tc) {
            int c = tc * 32 + lane;                 // d = c>>2, 16B sub-chunk
            int d = c >> 2;
            int o = c & 3;
            const char* gv = (const char*)(Vb + (size_t)d * KKEPT + j0 + o * 8);
            async_copy16(gv, lv + c * 16);
        }
    };
    stage(0, 0);
    wait_async();
#endif

    for (int j0 = 0; j0 < nkeys; j0 += 32) {
#if HAVE_ASYNC_LDS
        int buf = (j0 >> 5) & 1;
        if (j0 + 32 < nkeys) stage(j0 + 32, buf ^ 1);
        const _Float16* Kt = &Kl[buf][0];
        const _Float16* Vt = &Vl[buf][0];
#else
        if (j0 + 32 < nkeys) {
            __builtin_prefetch(Kb + (size_t)(j0 + 32) * HD, 0, 1);
            __builtin_prefetch(Vb + (size_t)(j0 + 32), 0, 1);
        }
#endif

        // ---- scores: S = Q K^T (two 16x16 tiles: keys j0.., j0+16..)
        v8f sA = {}, sB = {};
#pragma unroll
        for (int c = 0; c < 4; ++c) {
#if HAVE_ASYNC_LDS
            v16h kbA = load_tile16x32(Kt, HD, 0,  c * 32);
            v16h kbB = load_tile16x32(Kt, HD, 16, c * 32);
#else
            v16h kbA = load_tile16x32(Kb, HD, j0,      c * 32);
            v16h kbB = load_tile16x32(Kb, HD, j0 + 16, c * 32);
#endif
            sA = __builtin_amdgcn_wmma_f32_16x16x32_f16(
                false, qa[c], false, kbA, (short)0, sA, false, false);
            sB = __builtin_amdgcn_wmma_f32_16x16x32_f16(
                false, qa[c], false, kbB, (short)0, sB, false, false);
        }

        // ---- causal mask
        int kA = j0 + l16;
        int kB = j0 + 16 + l16;
        int pA = (kA < 4) ? kA : 1024 + kA;
        int pB = (kB < 4) ? kB : 1024 + kB;
#pragma unroll
        for (int r = 0; r < 8; ++r) {
            int qpos = NPAST + t * 16 + r + half * 8;
            if (pA > qpos) sA[r] = -3.0e38f;
            if (pB > qpos) sB[r] = -3.0e38f;
        }

        // ---- online softmax (rows live in 16-lane halves)
#pragma unroll
        for (int r = 0; r < 8; ++r) {
            float mx = fmaxf(sA[r], sB[r]);
            mx = fmaxf(mx, __shfl_xor(mx, 1, 32));
            mx = fmaxf(mx, __shfl_xor(mx, 2, 32));
            mx = fmaxf(mx, __shfl_xor(mx, 4, 32));
            mx = fmaxf(mx, __shfl_xor(mx, 8, 32));
            float mnew = fmaxf(m[r], mx);
            float corr = __expf(m[r] - mnew);
            float eA = __expf(sA[r] - mnew);
            float eB = __expf(sB[r] - mnew);
            float rsum = eA + eB;
            rsum += __shfl_xor(rsum, 1, 32);
            rsum += __shfl_xor(rsum, 2, 32);
            rsum += __shfl_xor(rsum, 4, 32);
            rsum += __shfl_xor(rsum, 8, 32);
            lsum[r] = lsum[r] * corr + rsum;
            m[r] = mnew;
            sA[r] = eA;
            sB[r] = eB;
#pragma unroll
            for (int c = 0; c < 8; ++c) O[c][r] *= corr;
        }

        // ---- stage P (16x32) through LDS: C-layout -> A-layout
#pragma unroll
        for (int r = 0; r < 8; ++r) {
            int row = r + half * 8;
            Plds[row * 32 + l16]      = (_Float16)sA[r];
            Plds[row * 32 + 16 + l16] = (_Float16)sB[r];
        }
        v16h pa = load_tile16x32(Plds, 32, 0, 0);  // same wave: LDS in-order

        // ---- O += P @ V
#pragma unroll
        for (int c = 0; c < 8; ++c) {
#if HAVE_ASYNC_LDS
            v16h vb = load_tile16x32(Vt, 32, c * 16, 0);
#else
            v16h vb = load_tile16x32(Vb, KKEPT, c * 16, j0);
#endif
            O[c] = __builtin_amdgcn_wmma_f32_16x16x32_f16(
                false, pa, false, vb, (short)0, O[c], false, false);
        }

#if HAVE_ASYNC_LDS
        wait_async();   // next buffer ready before the following iteration
#endif
    }

    // ---- epilogue: normalize, write f16 attn output [Q][H*D]
#pragma unroll
    for (int r = 0; r < 8; ++r) {
        int q = t * 16 + r + half * 8;
        float inv = 1.0f / lsum[r];
#pragma unroll
        for (int c = 0; c < 8; ++c) {
            a16[(size_t)q * (NH * HD) + h * HD + c * 16 + l16] =
                (_Float16)(O[c][r] * inv);
        }
    }
}

// ---------------------------------------------------------------------------
// Host-side orchestration.
// ---------------------------------------------------------------------------
extern "C" void kernel_launch(void* const* d_in, const int* in_sizes, int n_in,
                              void* d_out, int out_size, void* d_ws, size_t ws_size,
                              hipStream_t stream) {
    const float* hidden = (const float*)d_in[0];
    const float* Wq     = (const float*)d_in[1];
    const float* Wk     = (const float*)d_in[2];
    const float* Wv     = (const float*)d_in[3];
    const float* Wo     = (const float*)d_in[4];
    const float* past_k = (const float*)d_in[5];
    const float* past_v = (const float*)d_in[6];
    const float* cosT   = (const float*)d_in[7];
    const float* sinT   = (const float*)d_in[8];
    (void)in_sizes; (void)n_in; (void)out_size; (void)ws_size;
    float* out = (float*)d_out;

    char* ws = (char*)d_ws;
    _Float16* hid16   = (_Float16*)(ws + 0);           //  8 MiB
    _Float16* wqkv_t  = (_Float16*)(ws + 8388608);     // 48 MiB
    _Float16* wo_t    = (_Float16*)(ws + 58720256);    // 32 MiB
    float*    qkv_lin = (float*)   (ws + 92274688);    // 24 MiB
    _Float16* q16     = (_Float16*)(ws + 117440512);   //  8 MiB
    _Float16* k16     = (_Float16*)(ws + 125829120);   //  8 MiB
    _Float16* vt16    = (_Float16*)(ws + 134217728);   //  8 MiB
    _Float16* a16     = (_Float16*)(ws + 142606336);   //  8 MiB

    const int threads = 256;

    {   // conversions + weight transposes
        int n = NQ * DM;
        bantam_cvt_f16<<<(n + threads - 1) / threads, threads, 0, stream>>>(hidden, hid16, n);
    }
    {
        int n = DM * (NH * HD);
        bantam_tcvt_f16<<<(n + threads - 1) / threads, threads, 0, stream>>>(Wq, wqkv_t, DM, NH * HD);
    }
    {
        int n = DM * (NHKV * HD);
        bantam_tcvt_f16<<<(n + threads - 1) / threads, threads, 0, stream>>>(
            Wk, wqkv_t + (size_t)(NH * HD) * DM, DM, NHKV * HD);
        bantam_tcvt_f16<<<(n + threads - 1) / threads, threads, 0, stream>>>(
            Wv, wqkv_t + (size_t)(NH * HD + NHKV * HD) * DM, DM, NHKV * HD);
    }
    {
        int n = (NH * HD) * DM;
        bantam_tcvt_f16<<<(n + threads - 1) / threads, threads, 0, stream>>>(Wo, wo_t, NH * HD, DM);
    }

    {   // fused QKV projection
        dim3 grid(NQKV / 64, NQ / 32);
        bantam_gemm_wmma<<<grid, 32, 0, stream>>>(hid16, wqkv_t, qkv_lin, NQ, NQKV, DM);
    }

    {   // RoPE Q, gather+RoPE K, gather V (transposed)
        int n = NQ * NH * HD;
        bantam_rope_q<<<(n + threads - 1) / threads, threads, 0, stream>>>(qkv_lin, cosT, sinT, q16);
    }
    {
        int n = NHKV * KKEPT * HD;
        bantam_build_k<<<(n + threads - 1) / threads, threads, 0, stream>>>(
            qkv_lin, past_k, cosT, sinT, k16);
        bantam_build_v<<<(n + threads - 1) / threads, threads, 0, stream>>>(
            qkv_lin, past_v, vt16);
    }

    {   // flash attention
        dim3 grid(NH, NQ / 16);
        bantam_attn_wmma<<<grid, 32, 0, stream>>>(q16, k16, vt16, a16);
    }

    {   // output projection
        dim3 grid(DM / 64, NQ / 32);
        bantam_gemm_wmma<<<grid, 32, 0, stream>>>(a16, wo_t, out, NQ, DM, DM);
    }
}
